// Attention_71356586656326
// MI455X (gfx1250) — matrix operand order repaired and checked by reading, not compiled, not run on hardware
//
#include <hip/hip_runtime.h>
#include <hip/hip_bf16.h>

typedef _Float16 v8h  __attribute__((ext_vector_type(8)));
typedef _Float16 v16h __attribute__((ext_vector_type(16)));
typedef float    v8f  __attribute__((ext_vector_type(8)));
typedef int      i4v  __attribute__((vector_size(16)));

#define WMMA_F16(a, b, c) \
  __builtin_amdgcn_wmma_f32_16x16x32_f16(false, (a), false, (b), (short)0, (c), false, false)

// ---- CDNA5 async global->LDS copy (ASYNCcnt-tracked), with sync fallback ----
// clang signature: (v4i addrspace(1)*, v4i addrspace(3)*, imm offset, imm cpol)
#if __has_builtin(__builtin_amdgcn_global_load_async_to_lds_b128)
#define ASYNC_OK 1
#define ASYNC_CP(lds, glb)                                                  \
  __builtin_amdgcn_global_load_async_to_lds_b128(                           \
      (__attribute__((address_space(1))) i4v*)(glb),                        \
      (__attribute__((address_space(3))) i4v*)(lds), 0, 0)
#if __has_builtin(__builtin_amdgcn_s_wait_asynccnt)
#define WAIT_ASYNC(n) __builtin_amdgcn_s_wait_asynccnt(n)
#else
#define WAIT_ASYNC(n) asm volatile("s_wait_asynccnt %0" ::"i"(n) : "memory")
#endif
#else
#define ASYNC_OK 0
#endif

static __device__ __forceinline__ v16h cat16(v8h lo, v8h hi) {
  return __builtin_shufflevector(lo, hi, 0,1,2,3,4,5,6,7,8,9,10,11,12,13,14,15);
}
static __device__ __forceinline__ v8h ld8h(const _Float16* p) { return *(const v8h*)p; }
static __device__ __forceinline__ v8f vzero8() {
  v8f z = {0.f,0.f,0.f,0.f,0.f,0.f,0.f,0.f};
  return z;
}
static __device__ __forceinline__ v8h cvt8(float4 a, float4 b) {
  v8h r;
  r[0]=(_Float16)a.x; r[1]=(_Float16)a.y; r[2]=(_Float16)a.z; r[3]=(_Float16)a.w;
  r[4]=(_Float16)b.x; r[5]=(_Float16)b.y; r[6]=(_Float16)b.z; r[7]=(_Float16)b.w;
  return r;
}

// ---------------- prep kernels ----------------
__global__ void k_cvt_f16(const float* __restrict__ src, _Float16* __restrict__ dst, int n) {
  for (int i = blockIdx.x * blockDim.x + threadIdx.x; i < n; i += gridDim.x * blockDim.x)
    dst[i] = (_Float16)src[i];
}

// src[K][N] (row-major) -> dst[N][K] f16
__global__ void k_transpose_cvt(const float* __restrict__ src, _Float16* __restrict__ dst,
                                int K, int N) {
  long total = (long)K * N;
  for (long i = blockIdx.x * (long)blockDim.x + threadIdx.x; i < total;
       i += (long)gridDim.x * blockDim.x) {
    long n = i / K, k = i % K;
    dst[i] = (_Float16)src[k * (long)N + n];
  }
}

// ---------------- stage 2: fused QKV GEMM ----------------
// C[4096,3072] = xh[4096,1024] * wT^T + bias; scatter into q/k/vT head layouts (f16)
__global__ __launch_bounds__(256) void k_qkv_gemm(
    const _Float16* __restrict__ xh, const _Float16* __restrict__ wT,
    const float* __restrict__ bias,
    _Float16* __restrict__ q16, _Float16* __restrict__ k16, _Float16* __restrict__ vT16) {
  __shared__ _Float16 As[2][128 * 40];
  __shared__ _Float16 Bs[2][128 * 40];
  const int t = threadIdx.x, w = t >> 5, l = t & 31, lm = l & 15, lh = l >> 4;
  const int Mt = blockIdx.y * 128, Nt = blockIdx.x * 128;
  const int wm = w >> 1, wn = w & 1;

  v8f acc[2][4];
#pragma unroll
  for (int i = 0; i < 2; ++i)
#pragma unroll
    for (int j = 0; j < 4; ++j) acc[i][j] = vzero8();

  const int crow = t >> 1, cch = (t & 1) * 16;
  const _Float16* gA = xh + (size_t)(Mt + crow) * 1024 + cch;
  const _Float16* gB = wT + (size_t)(Nt + crow) * 1024 + cch;
  _Float16* const sA0 = &As[0][crow * 40 + cch];
  _Float16* const sA1 = &As[1][crow * 40 + cch];
  _Float16* const sB0 = &Bs[0][crow * 40 + cch];
  _Float16* const sB1 = &Bs[1][crow * 40 + cch];

#if ASYNC_OK
  ASYNC_CP(sA0, gA);  ASYNC_CP(sA0 + 8, gA + 8);
  ASYNC_CP(sB0, gB);  ASYNC_CP(sB0 + 8, gB + 8);
#endif

  for (int kc = 0; kc < 1024; kc += 32) {
    const int buf = (kc >> 5) & 1;
#if ASYNC_OK
    if (kc + 32 < 1024) {  // prefetch next tile into other buffer (overlaps WMMA)
      _Float16* nA = buf ? sA0 : sA1;
      _Float16* nB = buf ? sB0 : sB1;
      ASYNC_CP(nA, gA + kc + 32);  ASYNC_CP(nA + 8, gA + kc + 40);
      ASYNC_CP(nB, gB + kc + 32);  ASYNC_CP(nB + 8, gB + kc + 40);
      WAIT_ASYNC(4);  // own loads of current tile complete (in-order)
    } else {
      WAIT_ASYNC(0);
    }
#else
    {
      _Float16* dA = buf ? sA1 : sA0;
      _Float16* dB = buf ? sB1 : sB0;
      *(v8h*)(dA) = ld8h(gA + kc);  *(v8h*)(dA + 8) = ld8h(gA + kc + 8);
      *(v8h*)(dB) = ld8h(gB + kc);  *(v8h*)(dB + 8) = ld8h(gB + kc + 8);
    }
#endif
    __syncthreads();
    const _Float16* Ab = As[buf];
    const _Float16* Bb = Bs[buf];
    v16h af[2], bf[4];
#pragma unroll
    for (int i = 0; i < 2; ++i) {
      const _Float16* p = Ab + (wm * 32 + i * 16 + lm) * 40 + lh * 8;
      af[i] = cat16(ld8h(p), ld8h(p + 16));
    }
#pragma unroll
    for (int j = 0; j < 4; ++j) {
      const _Float16* p = Bb + (wn * 64 + j * 16 + lm) * 40 + lh * 8;
      bf[j] = cat16(ld8h(p), ld8h(p + 16));
    }
#pragma unroll
    for (int i = 0; i < 2; ++i)
#pragma unroll
      for (int j = 0; j < 4; ++j) acc[i][j] = WMMA_F16(af[i], bf[j], acc[i][j]);
    __syncthreads();
  }

  // epilogue: branch hoisted per tile (seg uniform across the 16-wide tile)
#pragma unroll
  for (int i = 0; i < 2; ++i)
#pragma unroll
    for (int j = 0; j < 4; ++j) {
      const int row0 = Mt + wm * 32 + i * 16 + (lh << 3);
      const int col = Nt + wn * 64 + j * 16 + lm;
      const float bv = bias[col];
      const int seg = col >> 10, c = col & 1023, h = c >> 6, hd = c & 63;
      const int b = row0 >> 10, s0 = row0 & 1023;  // 8-row group never crosses batch
      const size_t bhh = (size_t)b * 16 + h;
      if (seg == 2) {  // v: 8 consecutive s -> one 16B store per lane
        v8h pk;
#pragma unroll
        for (int r = 0; r < 8; ++r) pk[r] = (_Float16)(acc[i][j][r] + bv);
        *(v8h*)(vT16 + (bhh * 64 + hd) * 1024 + s0) = pk;
      } else {
        _Float16* dst = (seg == 0 ? q16 : k16) + (bhh * 1024 + s0) * 64 + hd;
#pragma unroll
        for (int r = 0; r < 8; ++r) dst[(size_t)r * 64] = (_Float16)(acc[i][j][r] + bv);
      }
    }
}

// ---------------- stage 3: fused scores + causal mask + softmax ----------------
// one WG = 64 query rows of one (b,h); 256KB f32 score block lives in LDS (CDNA5 320KB)
__global__ __launch_bounds__(256) void k_scores_softmax(
    const _Float16* __restrict__ q16, const _Float16* __restrict__ k16,
    const float* __restrict__ mask, float* __restrict__ wout) {
  extern __shared__ char smem[];
  float* sc = (float*)smem;                              // [64][1028] f32
  _Float16* qs = (_Float16*)(smem + 64 * 1028 * 4);      // [64][72]  f16
  const int t = threadIdx.x, w = t >> 5, l = t & 31, lm = l & 15, lh = l >> 4;
  const int blk = blockIdx.x;
  const int bh = blk >> 4, qb = (blk & 15) * 64, b = bh >> 4;

  {  // stage q tile [64][64]
    const int row = t >> 2, ch = (t & 3) * 16;
    const _Float16* g = q16 + ((size_t)bh * 1024 + qb + row) * 64 + ch;
    _Float16* s = qs + row * 72 + ch;
#if ASYNC_OK
    ASYNC_CP(s, g);  ASYNC_CP(s + 8, g + 8);
    WAIT_ASYNC(0);
#else
    *(v8h*)(s) = ld8h(g);  *(v8h*)(s + 8) = ld8h(g + 8);
#endif
  }
  __syncthreads();

  const int tm = w & 3, qrow_blk = tm * 16;
  v16h af[2];
#pragma unroll
  for (int kk = 0; kk < 2; ++kk) {
    const _Float16* p = qs + (qrow_blk + lm) * 72 + kk * 32 + lh * 8;
    af[kk] = cat16(ld8h(p), ld8h(p + 16));
  }
  const int qi_hi = qb + qrow_blk + 15;

  for (int tn = (w >> 2); tn < 64; tn += 2) {
    const int n0 = tn * 16;
    if (n0 > qi_hi) {  // tile entirely above causal diagonal
#pragma unroll
      for (int r = 0; r < 8; ++r)
        sc[(qrow_blk + r + (lh << 3)) * 1028 + n0 + lm] = -10000.0f;
      continue;
    }
    v8f acc = vzero8();
#pragma unroll
    for (int kk = 0; kk < 2; ++kk) {
      const _Float16* p = k16 + ((size_t)bh * 1024 + n0 + lm) * 64 + kk * 32 + lh * 8;
      v16h bf = cat16(ld8h(p), ld8h(p + 16));
      acc = WMMA_F16(af[kk], bf, acc);
    }
    const int j = n0 + lm;
#pragma unroll
    for (int r = 0; r < 8; ++r) {
      const int m = qrow_blk + r + (lh << 3);
      const int qi = qb + m;
      sc[m * 1028 + j] = (j <= qi) ? acc[r] * 0.125f : -10000.0f;
    }
  }
  __syncthreads();

  // row softmax: wave w owns rows [8w, 8w+8)
  for (int rr = 0; rr < 8; ++rr) {
    const int m = w * 8 + rr, qi = qb + m;
    float* row = sc + m * 1028;
    float mx = -3.0e38f;
    for (int it = 0; it < 32; ++it) {
      const int j = l + it * 32;
      const float v = row[j] + mask[(size_t)b * 1024 + j];
      row[j] = v;
      mx = fmaxf(mx, v);
    }
#pragma unroll
    for (int o = 16; o > 0; o >>= 1) mx = fmaxf(mx, __shfl_xor(mx, o));
    float sum = 0.f;
    for (int it = 0; it < 32; ++it) {
      const int j = l + it * 32;
      const float e = __expf(row[j] - mx);
      row[j] = e;
      sum += e;
    }
#pragma unroll
    for (int o = 16; o > 0; o >>= 1) sum += __shfl_xor(sum, o);
    const float inv = 1.0f / sum;
    float* dst = wout + ((size_t)bh * 1024 + qi) * 1024;
    for (int it = 0; it < 32; ++it) {
      const int j = l + it * 32;
      dst[j] = row[j] * inv;
    }
  }
}

// ---------------- stage 4: AV GEMM ----------------
// a'[b,s,h*64+hd] = sum_k w[bh,s,k] * v[bh,k,hd]; vT head parked in LDS (128KB)
__global__ __launch_bounds__(256) void k_av(
    const float* __restrict__ wprob, const _Float16* __restrict__ vT16,
    _Float16* __restrict__ ah) {
  extern __shared__ char smem[];
  _Float16* vs = (_Float16*)smem;  // [64][1032]
  const int t = threadIdx.x, w = t >> 5, l = t & 31, lm = l & 15, lh = l >> 4;
  const int bh = blockIdx.x >> 3, mb = (blockIdx.x & 7) * 128;
  const int b = bh >> 4, h = bh & 15;

  {  // stage vT[64][1024] into LDS
    const int row = t >> 2, cb = (t & 3) * 256;
    const _Float16* g = vT16 + ((size_t)bh * 64 + row) * 1024 + cb;
    _Float16* s = vs + row * 1032 + cb;
#if ASYNC_OK
#pragma unroll
    for (int i = 0; i < 32; ++i) ASYNC_CP(s + i * 8, g + i * 8);
    WAIT_ASYNC(0);
#else
#pragma unroll
    for (int i = 0; i < 32; ++i) *(v8h*)(s + i * 8) = ld8h(g + i * 8);
#endif
  }
  __syncthreads();

  const int m0 = mb + w * 16;
  v8f acc[4];
#pragma unroll
  for (int j = 0; j < 4; ++j) acc[j] = vzero8();
  const float* wrow = wprob + ((size_t)bh * 1024 + m0 + lm) * 1024;

  for (int kc = 0; kc < 1024; kc += 32) {
    const float* p = wrow + kc + lh * 8;
    __builtin_prefetch(p + 32, 0, 0);  // next K chunk of probabilities
    float4 f0 = *(const float4*)(p);
    float4 f1 = *(const float4*)(p + 4);
    float4 f2 = *(const float4*)(p + 16);
    float4 f3 = *(const float4*)(p + 20);
    v16h af = cat16(cvt8(f0, f1), cvt8(f2, f3));
#pragma unroll
    for (int j = 0; j < 4; ++j) {
      const _Float16* bp = vs + (j * 16 + lm) * 1032 + kc + lh * 8;
      v16h bf = cat16(ld8h(bp), ld8h(bp + 16));
      acc[j] = WMMA_F16(af, bf, acc[j]);
    }
  }
#pragma unroll
  for (int j = 0; j < 4; ++j)
#pragma unroll
    for (int r = 0; r < 8; ++r) {
      const int s = m0 + r + (lh << 3);
      const int col = h * 64 + j * 16 + lm;
      ah[((size_t)b * 1024 + s) * 1024 + col] = (_Float16)acc[j][r];
    }
}

// ---------------- stage 5: output projection ----------------
__global__ __launch_bounds__(256) void k_proj_gemm(
    const _Float16* __restrict__ ah, const _Float16* __restrict__ wT,
    const float* __restrict__ bias, float* __restrict__ out) {
  __shared__ _Float16 As[2][128 * 40];
  __shared__ _Float16 Bs[2][128 * 40];
  const int t = threadIdx.x, w = t >> 5, l = t & 31, lm = l & 15, lh = l >> 4;
  const int Mt = blockIdx.y * 128, Nt = blockIdx.x * 128;
  const int wm = w >> 1, wn = w & 1;

  v8f acc[2][4];
#pragma unroll
  for (int i = 0; i < 2; ++i)
#pragma unroll
    for (int j = 0; j < 4; ++j) acc[i][j] = vzero8();

  const int crow = t >> 1, cch = (t & 1) * 16;
  const _Float16* gA = ah + (size_t)(Mt + crow) * 1024 + cch;
  const _Float16* gB = wT + (size_t)(Nt + crow) * 1024 + cch;
  _Float16* const sA0 = &As[0][crow * 40 + cch];
  _Float16* const sA1 = &As[1][crow * 40 + cch];
  _Float16* const sB0 = &Bs[0][crow * 40 + cch];
  _Float16* const sB1 = &Bs[1][crow * 40 + cch];

#if ASYNC_OK
  ASYNC_CP(sA0, gA);  ASYNC_CP(sA0 + 8, gA + 8);
  ASYNC_CP(sB0, gB);  ASYNC_CP(sB0 + 8, gB + 8);
#endif

  for (int kc = 0; kc < 1024; kc += 32) {
    const int buf = (kc >> 5) & 1;
#if ASYNC_OK
    if (kc + 32 < 1024) {
      _Float16* nA = buf ? sA0 : sA1;
      _Float16* nB = buf ? sB0 : sB1;
      ASYNC_CP(nA, gA + kc + 32);  ASYNC_CP(nA + 8, gA + kc + 40);
      ASYNC_CP(nB, gB + kc + 32);  ASYNC_CP(nB + 8, gB + kc + 40);
      WAIT_ASYNC(4);
    } else {
      WAIT_ASYNC(0);
    }
#else
    {
      _Float16* dA = buf ? sA1 : sA0;
      _Float16* dB = buf ? sB1 : sB0;
      *(v8h*)(dA) = ld8h(gA + kc);  *(v8h*)(dA + 8) = ld8h(gA + kc + 8);
      *(v8h*)(dB) = ld8h(gB + kc);  *(v8h*)(dB + 8) = ld8h(gB + kc + 8);
    }
#endif
    __syncthreads();
    const _Float16* Ab = As[buf];
    const _Float16* Bb = Bs[buf];
    v16h af[2], bf[4];
#pragma unroll
    for (int i = 0; i < 2; ++i) {
      const _Float16* p = Ab + (wm * 32 + i * 16 + lm) * 40 + lh * 8;
      af[i] = cat16(ld8h(p), ld8h(p + 16));
    }
#pragma unroll
    for (int j = 0; j < 4; ++j) {
      const _Float16* p = Bb + (wn * 64 + j * 16 + lm) * 40 + lh * 8;
      bf[j] = cat16(ld8h(p), ld8h(p + 16));
    }
#pragma unroll
    for (int i = 0; i < 2; ++i)
#pragma unroll
      for (int j = 0; j < 4; ++j) acc[i][j] = WMMA_F16(af[i], bf[j], acc[i][j]);
    __syncthreads();
  }

#pragma unroll
  for (int i = 0; i < 2; ++i)
#pragma unroll
    for (int j = 0; j < 4; ++j) {
      const int row0 = Mt + wm * 32 + i * 16 + (lh << 3);
      const int col = Nt + wn * 64 + j * 16 + lm;
      const float bv = bias[col];
#pragma unroll
      for (int r = 0; r < 8; ++r)
        out[(size_t)(row0 + r) * 1024 + col] = acc[i][j][r] + bv;
    }
}

// ---------------- host launch ----------------
extern "C" void kernel_launch(void* const* d_in, const int* in_sizes, int n_in,
                              void* d_out, int out_size, void* d_ws, size_t ws_size,
                              hipStream_t stream) {
  const float* x      = (const float*)d_in[0];   // [4,1024,1024]
  const float* mask   = (const float*)d_in[1];   // [4,1,1,1024]
  const float* w_attn = (const float*)d_in[2];   // [1024,3072]
  const float* b_attn = (const float*)d_in[3];   // [1,3072]
  const float* w_proj = (const float*)d_in[4];   // [1024,1024]
  const float* b_proj = (const float*)d_in[5];   // [1,1024]

  float* out_a = (float*)d_out;                  // [4,1024,1024]
  float* out_w = (float*)d_out + (size_t)4 * 1024 * 1024;  // [4,16,1024,1024]

  char* ws = (char*)d_ws;
  _Float16* xh   = (_Float16*)(ws);                         //  8 MB [4096,1024]
  _Float16* wTa  = (_Float16*)(ws + 8388608);               //  6 MB [3072,1024]
  _Float16* wTp  = (_Float16*)(ws + 14680064);              //  2 MB [1024,1024]
  _Float16* q16  = (_Float16*)(ws + 16777216);              //  8 MB [64,1024,64]
  _Float16* k16  = (_Float16*)(ws + 25165824);              //  8 MB [64,1024,64]
  _Float16* vT16 = (_Float16*)(ws + 33554432);              //  8 MB [64,64,1024]
  _Float16* ah   = (_Float16*)(ws + 41943040);              //  8 MB [4096,1024]

  k_cvt_f16<<<4096, 256, 0, stream>>>(x, xh, 4 * 1024 * 1024);
  k_transpose_cvt<<<4096, 256, 0, stream>>>(w_attn, wTa, 1024, 3072);
  k_transpose_cvt<<<2048, 256, 0, stream>>>(w_proj, wTp, 1024, 1024);

  k_qkv_gemm<<<dim3(24, 32), 256, 0, stream>>>(xh, wTa, b_attn, q16, k16, vT16);

  const size_t smem2 = 64 * 1028 * 4 + 64 * 72 * 2;   // 272,384 B (<320KB WGP LDS)
  k_scores_softmax<<<1024, 256, smem2, stream>>>(q16, k16, mask, out_w);

  const size_t smem3 = 64 * 1032 * 2;                 // 132,096 B
  k_av<<<512, 256, smem3, stream>>>(out_w, vT16, ah);

  k_proj_gemm<<<dim3(8, 32), 256, 0, stream>>>(ah, wTp, b_proj, out_a);
}
